// SelfAttentionBlock_35777077575801
// MI455X (gfx1250) — matrix operand order repaired and checked by reading, not compiled, not run on hardware
//
#include <hip/hip_runtime.h>

// ---------------------------------------------------------------------------
// Types for CDNA5 WMMA (wave32)
// ---------------------------------------------------------------------------
typedef __attribute__((ext_vector_type(16))) __bf16     v16bf;
typedef __attribute__((ext_vector_type(8)))  float      v8f;
typedef __attribute__((ext_vector_type(8)))  unsigned   v8u;
typedef __attribute__((ext_vector_type(4)))  unsigned   v4u;
typedef __attribute__((ext_vector_type(8)))  int        v8i;
typedef __attribute__((ext_vector_type(4)))  int        v4i;

#define DEV __device__ __forceinline__

static constexpr int S_LEN   = 2304;   // 48*48
static constexpr int D_MODEL = 1024;
static constexpr int D_HEAD  = 64;
static constexpr int N_HEADS = 16;
static constexpr int COND    = 768;
static constexpr int BATCH   = 2;
static constexpr int M_ROWS  = BATCH * S_LEN;   // 4608

DEV unsigned short f2bf(float f) {
  unsigned u = __float_as_uint(f);
  unsigned r = u + 0x7FFFu + ((u >> 16) & 1u);   // round-to-nearest-even
  return (unsigned short)(r >> 16);
}

DEV v8f vzero() {
  v8f z;
#pragma unroll
  for (int i = 0; i < 8; ++i) z[i] = 0.f;
  return z;
}

DEV v8f wmma_bf16(v8u a, v8u b, v8f c) {
  return __builtin_amdgcn_wmma_f32_16x16x32_bf16(
      false, __builtin_bit_cast(v16bf, a),
      false, __builtin_bit_cast(v16bf, b),
      (short)0, c, false, false);
}

// A fragment: 16x32 bf16, row-major source, ld in elements.
DEV v8u load_a_frag(const unsigned short* p, int ld) {
  int lane = threadIdx.x & 31;
  int m = lane & 15, h = lane >> 4;
  const unsigned short* q = p + m * ld;
  v8u f;
#pragma unroll
  for (int v = 0; v < 8; ++v) {
    int k = (v < 4) ? (h * 8 + v * 2) : (16 + h * 8 + (v - 4) * 2);
    f[v] = *(const unsigned int*)(q + k);
  }
  return f;
}

// B fragment: 32x16 bf16 where B[k][n] = p[n*ld + k].
DEV v8u load_b_frag(const unsigned short* p, long ld) {
  int lane = threadIdx.x & 31;
  int n = lane & 15, h = lane >> 4;
  const unsigned short* q = p + (long)n * ld;
  v8u f;
#pragma unroll
  for (int v = 0; v < 8; ++v) f[v] = *(const unsigned int*)(q + ((v < 4) ? (h * 8 + v * 2) : (16 + h * 8 + (v - 4) * 2)));
  return f;
}

// ---------------------------------------------------------------------------
// TDM: 2D tensor_load_to_lds (D# built per CDNA5 ISA).
// Loads a tile of tile_d1 rows x tile_d0 elements (2-byte data) whose rows are
// stride0 elements apart in global memory, into contiguous LDS at lds_addr.
// ---------------------------------------------------------------------------
DEV void tdm_load_2d(const void* gptr, unsigned lds_addr,
                     unsigned tile_d0, unsigned tile_d1,
                     unsigned tensor_d0, unsigned tensor_d1,
                     unsigned long stride0_elems) {
  unsigned long ga = (unsigned long)gptr;
  v4u g0;
  g0[0] = 1u;                                   // count=1, user mode, no gather
  g0[1] = lds_addr;                             // LDS byte address
  g0[2] = (unsigned)(ga & 0xFFFFFFFFu);         // global_addr[31:0]
  g0[3] = (unsigned)((ga >> 32) & 0x01FFFFFFu)  // global_addr[56:32]
        | 0x80000000u;                          // type = 2 ("image")
  unsigned w0 = (1u << 16);                     // data_size = 1 -> 2 bytes
  unsigned w1 = (tensor_d0 & 0xFFFFu) << 16;    // tensor_dim0[15:0]
  unsigned w2 = ((tensor_d0 >> 16) & 0xFFFFu) | ((tensor_d1 & 0xFFFFu) << 16);
  unsigned w3 = ((tensor_d1 >> 16) & 0xFFFFu) | ((tile_d0 & 0xFFFFu) << 16);
  unsigned w4 = (tile_d1 & 0xFFFFu);            // tile_dim1 (tile_dim2 = 0)
  unsigned w5 = (unsigned)(stride0_elems & 0xFFFFFFFFu);
  unsigned w6 = (unsigned)((stride0_elems >> 32) & 0xFFFFu);
  unsigned w7 = 0u;
  v8i g1;
  g1[0] = (int)w0; g1[1] = (int)w1; g1[2] = (int)w2; g1[3] = (int)w3;
  g1[4] = (int)w4; g1[5] = (int)w5; g1[6] = (int)w6; g1[7] = (int)w7;
  v4i z4; z4[0] = 0; z4[1] = 0; z4[2] = 0; z4[3] = 0;
#if defined(__clang_major__) && (__clang_major__ >= 23)
  v8i z8;
#pragma unroll
  for (int i = 0; i < 8; ++i) z8[i] = 0;
  __builtin_amdgcn_tensor_load_to_lds(g0, g1, z4, z4, z8, 0);
#else
  __builtin_amdgcn_tensor_load_to_lds(g0, g1, z4, z4, 0);
#endif
}

DEV unsigned lds_addr_of(const void* p) {          // low 32 bits of generic addr
  return (unsigned)(unsigned long)p;               // = LDS byte offset (ISA 10.2)
}

// ---------------------------------------------------------------------------
// 0) f32 -> bf16 elementwise (weights)
// ---------------------------------------------------------------------------
__global__ void f2bf_kernel(const float* __restrict__ in,
                            unsigned short* __restrict__ out, long n) {
  long i = (long)blockIdx.x * blockDim.x + threadIdx.x;
  if (i < n) out[i] = f2bf(in[i]);
}

// ---------------------------------------------------------------------------
// 1) c[b][j] = 1 + cond[b] . norm_w[j]
// ---------------------------------------------------------------------------
__global__ void cond_kernel(const float* __restrict__ cond,
                            const float* __restrict__ norm_w,
                            float* __restrict__ cvec) {
  int idx = blockIdx.x * blockDim.x + threadIdx.x;
  if (idx >= BATCH * D_MODEL) return;
  int b = idx >> 10, j = idx & 1023;
  float acc = 1.0f;
  for (int t = 0; t < COND; ++t) acc += cond[b * COND + t] * norm_w[j * COND + t];
  cvec[idx] = acc;
}

// ---------------------------------------------------------------------------
// 2) RMSNorm + cond modulation -> bf16
// ---------------------------------------------------------------------------
__global__ __launch_bounds__(256) void rmsnorm_kernel(
    const float* __restrict__ x, const float* __restrict__ cvec,
    unsigned short* __restrict__ xn) {
  __shared__ float red[8];
  int row = blockIdx.x, tid = threadIdx.x;
  int b = row / S_LEN;
  const float* xr = x + (long)row * D_MODEL;
  float ss = 0.f;
  for (int i = tid; i < D_MODEL; i += 256) { float v = xr[i]; ss += v * v; }
#pragma unroll
  for (int m = 16; m >= 1; m >>= 1) ss += __shfl_xor(ss, m, 32);
  if ((tid & 31) == 0) red[tid >> 5] = ss;
  __syncthreads();
  if (tid == 0) {
    float t = 0.f;
#pragma unroll
    for (int i = 0; i < 8; ++i) t += red[i];
    red[0] = rsqrtf(t * (1.0f / D_MODEL) + 1e-6f);
  }
  __syncthreads();
  float inv = red[0];
  for (int i = tid; i < D_MODEL; i += 256)
    xn[(long)row * D_MODEL + i] = f2bf(xr[i] * cvec[b * D_MODEL + i] * inv);
}

// ---------------------------------------------------------------------------
// 3/6) GEMM: out[M][N] = A[M][K](bf16) @ W[N][K](bf16)^T (+skip), f32 out.
//     Block = 8 waves; wave tile 32(M) x 64(N) -> WG tile 64 x 256.
// ---------------------------------------------------------------------------
__global__ __launch_bounds__(256) void gemm_bf16_wmma(
    const unsigned short* __restrict__ A, const unsigned short* __restrict__ W,
    const float* __restrict__ skip, float* __restrict__ out,
    int M, int N, int K) {
  int wave = threadIdx.x >> 5, lane = threadIdx.x & 31;
  int row0 = blockIdx.x * 64 + (wave >> 2) * 32;
  int col0 = blockIdx.y * 256 + (wave & 3) * 64;
  v8f acc[2][4];
#pragma unroll
  for (int i = 0; i < 2; ++i)
#pragma unroll
    for (int t = 0; t < 4; ++t) acc[i][t] = vzero();

  for (int kk = 0; kk < K; kk += 32) {
    v8u a0 = load_a_frag(A + (long)row0 * K + kk, K);
    v8u a1 = load_a_frag(A + (long)(row0 + 16) * K + kk, K);
#pragma unroll
    for (int t = 0; t < 4; ++t) {
      v8u b = load_b_frag(W + (long)(col0 + t * 16) * K + kk, K);
      acc[0][t] = wmma_bf16(a0, b, acc[0][t]);
      acc[1][t] = wmma_bf16(a1, b, acc[1][t]);
    }
  }
  int hh = lane >> 4, cc = lane & 15;
#pragma unroll
  for (int i = 0; i < 2; ++i)
#pragma unroll
    for (int t = 0; t < 4; ++t)
#pragma unroll
      for (int r = 0; r < 8; ++r) {
        long m = row0 + i * 16 + hh * 8 + r;
        long n = col0 + t * 16 + cc;
        float v = acc[i][t][r];
        if (skip) v += skip[m * N + n];
        out[m * N + n] = v;
      }
}

// ---------------------------------------------------------------------------
// 4) q/k norm, axial rotary, head-split; q,k as [b*h][S][64] bf16,
//    v transposed as [b*h][64][S] bf16.
// ---------------------------------------------------------------------------
__global__ void prep_kernel(const float* __restrict__ qkv,
                            const float* __restrict__ pos,
                            const float* __restrict__ scale,
                            const float* __restrict__ freqs,
                            unsigned short* __restrict__ qh,
                            unsigned short* __restrict__ kh,
                            unsigned short* __restrict__ vT) {
  int idx = blockIdx.x * blockDim.x + threadIdx.x;
  if (idx >= BATCH * N_HEADS * S_LEN) return;
  int b = idx / (N_HEADS * S_LEN);
  int r = idx % (N_HEADS * S_LEN);
  int h = r / S_LEN, s = r % S_LEN;

  long base = ((long)(b * S_LEN + s)) * (3 * D_MODEL);
  const float* qp = qkv + base + 0 * D_MODEL + h * D_HEAD;
  const float* kp = qkv + base + 1 * D_MODEL + h * D_HEAD;
  const float* vp = qkv + base + 2 * D_MODEL + h * D_HEAD;

  float q[D_HEAD], k[D_HEAD];
  float sq = 0.f, sk = 0.f;
#pragma unroll 8
  for (int e = 0; e < D_HEAD; ++e) {
    q[e] = qp[e]; sq += q[e] * q[e];
    k[e] = kp[e]; sk += k[e] * k[e];
  }
  float qs = sqrtf(scale[h]) * rsqrtf(sq + 1e-6f);
  float ks = sqrtf(scale[h]) * rsqrtf(sk + 1e-6f);

  float p0 = pos[(long)(b * S_LEN + s) * 2 + 0];
  float p1 = pos[(long)(b * S_LEN + s) * 2 + 1];
  float qo[D_HEAD], ko[D_HEAD];
#pragma unroll
  for (int j = 0; j < 16; ++j) {
    float th = (j < 8) ? p0 * freqs[h * 8 + j] : p1 * freqs[h * 8 + (j - 8)];
    float c = cosf(th), sn = sinf(th);
    float q1 = q[j] * qs, q2 = q[16 + j] * qs;
    qo[j] = q1 * c - q2 * sn;  qo[16 + j] = q2 * c + q1 * sn;
    float k1 = k[j] * ks, k2 = k[16 + j] * ks;
    ko[j] = k1 * c - k2 * sn;  ko[16 + j] = k2 * c + k1 * sn;
  }
#pragma unroll
  for (int e = 32; e < D_HEAD; ++e) { qo[e] = q[e] * qs; ko[e] = k[e] * ks; }

  long ob = ((long)(b * N_HEADS + h) * S_LEN + s) * D_HEAD;
#pragma unroll 8
  for (int e = 0; e < D_HEAD; ++e) { qh[ob + e] = f2bf(qo[e]); kh[ob + e] = f2bf(ko[e]); }
  long vb = (long)(b * N_HEADS + h) * D_HEAD * S_LEN;
#pragma unroll 8
  for (int e = 0; e < D_HEAD; ++e) vT[vb + (long)e * S_LEN + s] = f2bf(vp[e]);
}

// ---------------------------------------------------------------------------
// 5) Flash attention. 8 waves x 16 q-rows = 128 q rows per WG.
//    K/V 32-key tiles are staged in LDS by wave 0 via TDM tensor_load_to_lds
//    (double-buffered, TENSORcnt + barriers); all waves consume from LDS.
// ---------------------------------------------------------------------------
__global__ __launch_bounds__(256) void attn_kernel(
    const unsigned short* __restrict__ qh, const unsigned short* __restrict__ kh,
    const unsigned short* __restrict__ vT, unsigned short* __restrict__ obf) {
  __shared__ __align__(16) unsigned short Ktile[2][32][64];  // [key][e]
  __shared__ __align__(16) unsigned short Vtile[2][64][32];  // [e-col][key]
  __shared__ __align__(16) unsigned short pl[8][16][40];     // per-wave P tile
  int wave = threadIdx.x >> 5, lane = threadIdx.x & 31;
  int bh = blockIdx.y;
  int b = bh >> 4, h = bh & 15;
  int qbase = blockIdx.x * 128 + wave * 16;

  const unsigned short* Q  = qh + (long)bh * S_LEN * D_HEAD;
  const unsigned short* Kp = kh + (long)bh * S_LEN * D_HEAD;
  const unsigned short* Vt = vT + (long)bh * D_HEAD * S_LEN;

  v8u aq0 = load_a_frag(Q + (long)qbase * D_HEAD + 0, D_HEAD);
  v8u aq1 = load_a_frag(Q + (long)qbase * D_HEAD + 32, D_HEAD);

  v8f acc[4];
#pragma unroll
  for (int t = 0; t < 4; ++t) acc[t] = vzero();
  float mrow[8], lrow[8];
#pragma unroll
  for (int r = 0; r < 8; ++r) { mrow[r] = -1e30f; lrow[r] = 0.f; }

  const int NT = S_LEN / 32;   // 72 key tiles
  if (wave == 0) {
    // K tile: 32 rows x 64 elems, rows stride 64 (contiguous 4KB block)
    tdm_load_2d(Kp, lds_addr_of(&Ktile[0][0][0]), 64, 32, 64, S_LEN, 64);
    // V tile: 64 rows x 32 elems, rows stride S_LEN
    tdm_load_2d(Vt, lds_addr_of(&Vtile[0][0][0]), 32, 64, S_LEN, 64, S_LEN);
  }

  for (int j = 0; j < NT; ++j) {
    int kt = j * 32, buf = j & 1;
    if (wave == 0) __builtin_amdgcn_s_wait_tensorcnt(0);
    __syncthreads();
    if (wave == 0 && (j + 1) < NT) {
      int nk = kt + 32, nb = buf ^ 1;
      tdm_load_2d(Kp + (long)nk * D_HEAD, lds_addr_of(&Ktile[nb][0][0]),
                  64, 32, 64, S_LEN, 64);
      tdm_load_2d(Vt + nk, lds_addr_of(&Vtile[nb][0][0]),
                  32, 64, S_LEN, 64, S_LEN);
    }

    // logits = q @ k^T for keys [kt, kt+32)  (K from LDS)
    v8f c0 = vzero(), c1 = vzero();
    c0 = wmma_bf16(aq0, load_b_frag(&Ktile[buf][0][0] + 0, 64), c0);
    c0 = wmma_bf16(aq1, load_b_frag(&Ktile[buf][0][0] + 32, 64), c0);
    c1 = wmma_bf16(aq0, load_b_frag(&Ktile[buf][16][0] + 0, 64), c1);
    c1 = wmma_bf16(aq1, load_b_frag(&Ktile[buf][16][0] + 32, 64), c1);

    // online softmax: rows in VGPR index r (row = half*8 + r), cols in lanes
    float tmax[8];
#pragma unroll
    for (int r = 0; r < 8; ++r) tmax[r] = fmaxf(c0[r], c1[r]);
#pragma unroll
    for (int m = 1; m < 16; m <<= 1)
#pragma unroll
      for (int r = 0; r < 8; ++r) tmax[r] = fmaxf(tmax[r], __shfl_xor(tmax[r], m, 32));

    float alpha[8], ps0[8], ps1[8], rsum[8];
#pragma unroll
    for (int r = 0; r < 8; ++r) {
      float nm = fmaxf(mrow[r], tmax[r]);
      alpha[r] = __expf(mrow[r] - nm);
      mrow[r]  = nm;
      ps0[r] = __expf(c0[r] - nm);
      ps1[r] = __expf(c1[r] - nm);
      rsum[r] = ps0[r] + ps1[r];
    }
#pragma unroll
    for (int m = 1; m < 16; m <<= 1)
#pragma unroll
      for (int r = 0; r < 8; ++r) rsum[r] += __shfl_xor(rsum[r], m, 32);
#pragma unroll
    for (int r = 0; r < 8; ++r) lrow[r] = lrow[r] * alpha[r] + rsum[r];
#pragma unroll
    for (int t = 0; t < 4; ++t)
#pragma unroll
      for (int r = 0; r < 8; ++r) acc[t][r] *= alpha[r];

    // C-layout -> A-fragment layout via per-wave LDS tile
    unsigned short (*pp)[40] = pl[wave];
    int rowh = lane >> 4, col = lane & 15;
#pragma unroll
    for (int r = 0; r < 8; ++r) {
      pp[rowh * 8 + r][col]      = f2bf(ps0[r]);
      pp[rowh * 8 + r][col + 16] = f2bf(ps1[r]);
    }
    asm volatile("s_wait_dscnt 0" ::: "memory");
    v8u ap = load_a_frag(&pp[0][0], 40);

    // o += P @ V   (V from LDS: B[k][n] = Vtile[t*16+n][k])
#pragma unroll
    for (int t = 0; t < 4; ++t) {
      v8u bv = load_b_frag(&Vtile[buf][t * 16][0], 32);
      acc[t] = wmma_bf16(ap, bv, acc[t]);
    }
    __syncthreads();   // all waves done reading buf before it is overwritten
  }

  float inv[8];
#pragma unroll
  for (int r = 0; r < 8; ++r) inv[r] = 1.0f / lrow[r];
  int hh = lane >> 4, cc = lane & 15;
#pragma unroll
  for (int t = 0; t < 4; ++t)
#pragma unroll
    for (int r = 0; r < 8; ++r) {
      long m = (long)b * S_LEN + qbase + hh * 8 + r;
      long n = (long)h * D_HEAD + t * 16 + cc;
      obf[m * D_MODEL + n] = f2bf(acc[t][r] * inv[r]);
    }
}

// ---------------------------------------------------------------------------
// Launch
// ---------------------------------------------------------------------------
extern "C" void kernel_launch(void* const* d_in, const int* in_sizes, int n_in,
                              void* d_out, int out_size, void* d_ws, size_t ws_size,
                              hipStream_t stream) {
  const float* x      = (const float*)d_in[0];
  const float* pos    = (const float*)d_in[1];
  const float* cond   = (const float*)d_in[2];
  const float* norm_w = (const float*)d_in[3];
  const float* qkv_w  = (const float*)d_in[4];
  const float* out_w  = (const float*)d_in[5];
  const float* scale  = (const float*)d_in[6];
  const float* freqs  = (const float*)d_in[7];
  float* out = (float*)d_out;

  size_t off = 0;
  auto alloc = [&](size_t bytes) -> void* {
    void* p = (char*)d_ws + off;
    off += (bytes + 255) & ~(size_t)255;
    return p;
  };
  unsigned short* qkvw_bf = (unsigned short*)alloc((size_t)3 * D_MODEL * D_MODEL * 2);
  unsigned short* outw_bf = (unsigned short*)alloc((size_t)D_MODEL * D_MODEL * 2);
  float*          cvec    = (float*)alloc((size_t)BATCH * D_MODEL * 4);
  unsigned short* xn      = (unsigned short*)alloc((size_t)M_ROWS * D_MODEL * 2);
  float*          qkv     = (float*)alloc((size_t)M_ROWS * 3 * D_MODEL * 4);
  unsigned short* qh_     = (unsigned short*)alloc((size_t)BATCH * N_HEADS * S_LEN * D_HEAD * 2);
  unsigned short* kh_     = (unsigned short*)alloc((size_t)BATCH * N_HEADS * S_LEN * D_HEAD * 2);
  unsigned short* vT_     = (unsigned short*)alloc((size_t)BATCH * N_HEADS * S_LEN * D_HEAD * 2);
  unsigned short* obf     = (unsigned short*)alloc((size_t)M_ROWS * D_MODEL * 2);

  long nqw = (long)3 * D_MODEL * D_MODEL;
  f2bf_kernel<<<(nqw + 255) / 256, 256, 0, stream>>>(qkv_w, qkvw_bf, nqw);
  long now = (long)D_MODEL * D_MODEL;
  f2bf_kernel<<<(now + 255) / 256, 256, 0, stream>>>(out_w, outw_bf, now);

  cond_kernel<<<(BATCH * D_MODEL + 255) / 256, 256, 0, stream>>>(cond, norm_w, cvec);
  rmsnorm_kernel<<<M_ROWS, 256, 0, stream>>>(x, cvec, xn);

  gemm_bf16_wmma<<<dim3(M_ROWS / 64, (3 * D_MODEL) / 256), 256, 0, stream>>>(
      xn, qkvw_bf, nullptr, qkv, M_ROWS, 3 * D_MODEL, D_MODEL);

  int nprep = BATCH * N_HEADS * S_LEN;
  prep_kernel<<<(nprep + 255) / 256, 256, 0, stream>>>(qkv, pos, scale, freqs,
                                                       qh_, kh_, vT_);

  attn_kernel<<<dim3(S_LEN / 128, BATCH * N_HEADS), 256, 0, stream>>>(qh_, kh_, vT_, obf);

  gemm_bf16_wmma<<<dim3(M_ROWS / 64, D_MODEL / 256), 256, 0, stream>>>(
      obf, outw_bf, x, out, M_ROWS, D_MODEL, D_MODEL);
}
